// KernelDecoder_5317169512774
// MI455X (gfx1250) — matrix operand order repaired and checked by reading, not compiled, not run on hardware
//
#include <hip/hip_runtime.h>
#include <hip/hip_bf16.h>

typedef __attribute__((ext_vector_type(16))) __bf16          v16bf;
typedef __attribute__((ext_vector_type(8)))  float           v8f;
typedef __attribute__((ext_vector_type(16))) unsigned short  v16us;
typedef __attribute__((ext_vector_type(8)))  unsigned short  v8us;

#define B_   2
#define N_   4096
#define D_   128
#define T_   384
#define K_   9
#define NL   32
#define C1   259          // 2D+3
#define C1P  288          // padded to 9*32 for bf16 WMMA K-steps
#define H_   256          // 2D hidden
#define COLS (B_*N_*8)    // 65536 MLP columns

__device__ __forceinline__ unsigned short f2bf(float f) {
    unsigned u = __float_as_uint(f);
    u += 0x7FFFu + ((u >> 16) & 1u);          // round-to-nearest-even
    return (unsigned short)(u >> 16);
}

__device__ __forceinline__ v8f wmma_bf16(v16us a, v16us b, v8f c) {
    return __builtin_amdgcn_wmma_f32_16x16x32_bf16(
        false, __builtin_bit_cast(v16bf, a),
        false, __builtin_bit_cast(v16bf, b),
        (short)0, c, false, false);
}

// ---------------------------------------------------------------------------
// Pack a row-major fp32 weight matrix [mtiles*16 x K] into bf16 WMMA
// A-fragment order: out[((mtile*KS + ks)*32 + lane)*16 + e], zero-pad K.
//   row = mtile*16 + (lane&15)
//   k   = ks*32 + e + ((lane>>4)<<4)   (contiguous k order, matches B loads)
// ---------------------------------------------------------------------------
__global__ __launch_bounds__(256) void pack_w_kernel(
        const float* __restrict__ W, unsigned short* __restrict__ out,
        int Kdim, int KS)
{
    const int gid  = blockIdx.x * 256 + threadIdx.x;   // one element each
    const int e    = gid & 15;
    const int lane = (gid >> 4) & 31;
    const int ks   = (gid >> 9) % KS;
    const int mt   = (gid >> 9) / KS;
    const int row  = mt * 16 + (lane & 15);
    const int ka   = ks * 32 + e + ((lane >> 4) << 4);
    out[gid] = (ka < Kdim) ? f2bf(W[row * Kdim + ka]) : (unsigned short)0;
}

// SFt[(b*N+n)*128 + d] = bf16(SF[b][d][n])  (coalesced read, scattered write)
__global__ __launch_bounds__(256) void pack_sft_kernel(
        const float* __restrict__ SF, unsigned short* __restrict__ SFt)
{
    const int lin = blockIdx.x * 256 + threadIdx.x;    // over B*D*N
    const int b = lin / (D_ * N_);
    const int r = lin - b * (D_ * N_);
    const int d = r >> 12, n = r & (N_ - 1);
    SFt[(b * N_ + n) * D_ + d] = f2bf(SF[lin]);
}

// ---------------------------------------------------------------------------
// Kernel A: radius ball query. One wave per query point; batch points in LDS.
// Ballot + prefix popcount appends hits in index order == reference's
// sort-by-index / first-32 selection. Shadow = N_.
// ---------------------------------------------------------------------------
__global__ __launch_bounds__(256) void ball_query_kernel(
        const float* __restrict__ SP, int* __restrict__ gidx)
{
    __shared__ float pts[N_ * 3];
    __shared__ int   sel[8 * NL];
    const int t    = threadIdx.x;
    const int wave = t >> 5;
    const int lane = t & 31;
    const int qb   = blockIdx.x;          // 0..1023, 8 queries per block
    const int b    = (qb * 8) / N_;       // whole block is one batch (512 | N_/8)
    const float* sp = SP + b * (3 * N_);
    #pragma unroll 4
    for (int j = 0; j < 48; ++j) {        // 12288 floats / 256 threads
        int lin = t + 256 * j;
        int c = lin >> 12, idx = lin & (N_ - 1);
        pts[idx * 3 + c] = sp[lin];
    }
    sel[wave * NL + lane] = N_;           // shadow fill
    __syncthreads();

    const int n  = qb * 8 + wave - b * N_;
    const float qx = pts[n * 3 + 0], qy = pts[n * 3 + 1], qz = pts[n * 3 + 2];
    const float snq = qx * qx + qy * qy + qz * qz;
    int cnt = 0;
    for (int base = 0; base < N_ && cnt < NL; base += 32) {
        int m = base + lane;
        float x = pts[m * 3 + 0], y = pts[m * 3 + 1], z = pts[m * 3 + 2];
        float sm  = x * x + y * y + z * z;
        float dot = x * qx + y * qy + z * qz;
        float sq  = snq + sm - 2.0f * dot;            // reference's expansion
        bool pred = (sq <= 0.09f);                    // CONV_RADIUS^2
        unsigned mask = __builtin_amdgcn_ballot_w32(pred);
        int pos = cnt + __popc(mask & ((1u << lane) - 1u));
        if (pred && pos < NL) sel[wave * NL + pos] = m;
        cnt += __popc(mask);
    }
    __syncthreads();
    gidx[(b * N_ + n) * NL + lane] = sel[wave * NL + lane];
}

// ---------------------------------------------------------------------------
// Kernel B: begin_conv: pre[b][n][d] = w_begin @ sf + b_begin. bf16 WMMA.
// 4-way M-blocked: wave = (ctile, mgroup of 4 m-tiles). 512 x 2 = 1024 waves.
// ---------------------------------------------------------------------------
__global__ __launch_bounds__(256) void begin_conv_kernel(
        const unsigned short* __restrict__ SFt, const unsigned short* __restrict__ WbS,
        const float* __restrict__ bias0, float* __restrict__ pre)
{
    const int t = threadIdx.x, lane = t & 31, hi = lane >> 4, ln = lane & 15;
    const int gw    = blockIdx.x * 8 + (t >> 5);
    const int mg    = gw & 1, ctile = gw >> 1;
    const int col   = ctile * 16 + ln;            // col = b*N + n
    v8f acc[4] = {};
    for (int ks = 0; ks < 4; ++ks) {
        v16us bu = *(const v16us*)(SFt + col * D_ + ks * 32 + (hi << 4));
        #pragma unroll
        for (int j = 0; j < 4; ++j) {
            const int mtile = mg * 4 + j;
            v16us au = *(const v16us*)(WbS + (((mtile * 4 + ks) << 5) + lane) * 16);
            acc[j] = wmma_bf16(au, bu, acc[j]);
        }
    }
    #pragma unroll
    for (int j = 0; j < 4; ++j) {
        const int mb = (mg * 4 + j) * 16 + (hi << 3);
        #pragma unroll
        for (int r = 0; r < 8; ++r)
            pre[col * D_ + mb + r] = acc[j][r] + bias0[mb + r];
    }
}

// ---------------------------------------------------------------------------
// Kernel C: per point, influence weights iw[k][m], wf[d][k] = sum_m iw*nf,
// assemble 259-ch (288-padded) bf16 column matrix X[col][ch].
// Raw-view kp for distances; proper-transpose qp for the concat — as in ref.
// ---------------------------------------------------------------------------
__global__ __launch_bounds__(256) void assemble_kernel(
        const float* __restrict__ SP, const unsigned short* __restrict__ SFt,
        const float* __restrict__ QP, const int* __restrict__ gidx,
        const float* __restrict__ pre, unsigned short* __restrict__ X)
{
    __shared__ float iw[8 * NL];       // [k][m]
    __shared__ float nf[NL * D_];      // [m][d], 16 KB
    __shared__ int   gsh[NL];
    const int t  = threadIdx.x;
    const int bn = blockIdx.x;                 // 0..8191 == col/8
    const int b  = bn >> 12, n = bn & (N_ - 1);
    if (t < NL) gsh[t] = gidx[bn * NL + t];
    __syncthreads();

    {   // stage 1: iw (256 threads = 8 kp x 32 neighbors)
        int k = t >> 5, m = t & 31;
        int idx = gsh[m];
        float w = 0.0f;
        if (idx < N_) {
            float sq = 0.0f;
            #pragma unroll
            for (int c = 0; c < 3; ++c) {
                float rel = SP[b * 3 * N_ + c * N_ + idx] - SP[b * 3 * N_ + c * N_ + n];
                float kp  = QP[c * K_ + (k + 1)];      // raw row-major reinterpret
                float dd  = rel - kp;
                sq += dd * dd;
            }
            w = 1.0f - sqrtf(sq) * (1.0f / 0.15f);
            w = w > 0.0f ? w : 0.0f;
        }
        iw[k * NL + m] = w;
    }
    {   // stage 2: gather neighbor features (shadow -> 0)
        int m = t >> 3, d0 = (t & 7) << 4;
        int idx = gsh[m];
        if (idx < N_) {
            const float* src = pre + (b * N_ + idx) * D_ + d0;
            #pragma unroll
            for (int j = 0; j < 16; ++j) nf[m * D_ + d0 + j] = src[j];
        } else {
            #pragma unroll
            for (int j = 0; j < 16; ++j) nf[m * D_ + d0 + j] = 0.0f;
        }
    }
    __syncthreads();

    const int colbase = bn * 8;
    // stage 3: wf -> channels [0,128)
    #pragma unroll
    for (int r = 0; r < 4; ++r) {
        int lin = t + 256 * r;
        int d = lin & 127, k = lin >> 7;
        float acc = 0.0f;
        #pragma unroll 8
        for (int m = 0; m < NL; ++m) acc += iw[k * NL + m] * nf[m * D_ + d];
        X[(colbase + k) * C1P + d] = f2bf(acc);
    }
    // stage 4: channels [128,288): qp (proper transpose), sf (pre-converted), pad
    #pragma unroll
    for (int r = 0; r < 5; ++r) {
        int lin = t + 256 * r;                 // 0..1279 = 8 k * 160 ch
        int k = lin / 160, c = lin % 160;
        unsigned short v;
        if (c < 3)        v = f2bf(QP[(k + 1) * 3 + c]);
        else if (c < 131) v = SFt[bn * D_ + (c - 3)];
        else              v = 0;
        X[(colbase + k) * C1P + 128 + c] = v;
    }
}

// ---------------------------------------------------------------------------
// Kernel D: layer1  Y = relu(W1[256x259] X + b1).
// 4-way M-blocked: wave = (ctile, mgroup). 4096 x 4 = 16384 waves.
// Per k-step: 1 shared B load + 4 A loads + 4 independent WMMAs.
// ---------------------------------------------------------------------------
__global__ __launch_bounds__(256) void mlp1_kernel(
        const unsigned short* __restrict__ X, const unsigned short* __restrict__ W1s,
        const float* __restrict__ b1, unsigned short* __restrict__ Y)
{
    const int t = threadIdx.x, lane = t & 31, hi = lane >> 4, ln = lane & 15;
    const int gw    = blockIdx.x * 8 + (t >> 5);
    const int mg    = gw & 3, ctile = gw >> 2;
    const int col   = ctile * 16 + ln;
    v8f acc[4] = {};
    for (int ks = 0; ks < 9; ++ks) {
        v16us bu = *(const v16us*)(X + col * C1P + ks * 32 + (hi << 4));
        #pragma unroll
        for (int j = 0; j < 4; ++j) {
            const int mtile = mg * 4 + j;
            v16us au = *(const v16us*)(W1s + (((mtile * 9 + ks) << 5) + lane) * 16);
            acc[j] = wmma_bf16(au, bu, acc[j]);
        }
    }
    #pragma unroll
    for (int j = 0; j < 4; ++j) {
        const int mb = (mg * 4 + j) * 16 + (hi << 3);
        v8us o;
        #pragma unroll
        for (int r = 0; r < 8; ++r) {
            float y = acc[j][r] + b1[mb + r];
            o[r] = f2bf(y > 0.0f ? y : 0.0f);
        }
        *(v8us*)(Y + col * H_ + mb) = o;
    }
}

// ---------------------------------------------------------------------------
// Kernel E: layer2  out = relu(W2[384x256] Y + b2) -> [B,384,N,8] fp32.
// 4-way M-blocked: wave = (ctile, mgroup of 4). 4096 x 6 = 24576 waves.
// ---------------------------------------------------------------------------
__global__ __launch_bounds__(256) void mlp2_kernel(
        const unsigned short* __restrict__ Y, const unsigned short* __restrict__ W2s,
        const float* __restrict__ b2, float* __restrict__ out)
{
    const int t = threadIdx.x, lane = t & 31, hi = lane >> 4, ln = lane & 15;
    const int gw    = blockIdx.x * 8 + (t >> 5);
    const int mg    = gw % 6, ctile = gw / 6;
    const int col   = ctile * 16 + ln;
    v8f acc[4] = {};
    for (int ks = 0; ks < 8; ++ks) {
        v16us bu = *(const v16us*)(Y + col * H_ + ks * 32 + (hi << 4));
        #pragma unroll
        for (int j = 0; j < 4; ++j) {
            const int mtile = mg * 4 + j;
            v16us au = *(const v16us*)(W2s + (((mtile * 8 + ks) << 5) + lane) * 16);
            acc[j] = wmma_bf16(au, bu, acc[j]);
        }
    }
    const int cb = col >> 15, rem = col & 32767;
    const int n = rem >> 3, kk = rem & 7;
    float* obase = out + cb * (T_ * N_ * 8) + n * 8 + kk;
    #pragma unroll
    for (int j = 0; j < 4; ++j) {
        const int tb = (mg * 4 + j) * 16 + (hi << 3);
        #pragma unroll
        for (int r = 0; r < 8; ++r) {
            float z = acc[j][r] + b2[tb + r];
            obase[(tb + r) * (N_ * 8)] = z > 0.0f ? z : 0.0f;
        }
    }
}

// ---------------------------------------------------------------------------
extern "C" void kernel_launch(void* const* d_in, const int* in_sizes, int n_in,
                              void* d_out, int out_size, void* d_ws, size_t ws_size,
                              hipStream_t stream) {
    const float* SP = (const float*)d_in[0];   // [B,3,N]
    const float* SF = (const float*)d_in[1];   // [B,D,N]
    const float* QP = (const float*)d_in[2];   // [K,3]
    const float* Wb = (const float*)d_in[3];   // [D,D]
    const float* bb = (const float*)d_in[4];   // [D]
    const float* W1 = (const float*)d_in[5];   // [256,259]
    const float* b1 = (const float*)d_in[6];   // [256]
    const float* W2 = (const float*)d_in[7];   // [384,256]
    const float* b2 = (const float*)d_in[8];   // [384]
    (void)in_sizes; (void)n_in; (void)out_size; (void)ws_size;

    char* ws = (char*)d_ws;
    int*            gidx = (int*)(ws);                                  // 1 MiB
    float*          pre  = (float*)(ws + (size_t)(1  << 20));           // 4 MiB
    unsigned short* X    = (unsigned short*)(ws + (size_t)(5  << 20));  // 36 MiB
    unsigned short* Yb   = (unsigned short*)(ws + (size_t)(43 << 20));  // 32 MiB
    char* wsw = ws + (size_t)(76 << 20);
    unsigned short* WbS  = (unsigned short*)(wsw);                      //  32 KiB
    unsigned short* W1s  = (unsigned short*)(wsw + (64  << 10));        // 144 KiB
    unsigned short* W2s  = (unsigned short*)(wsw + (256 << 10));        // 192 KiB
    unsigned short* SFt  = (unsigned short*)(wsw + (512 << 10));        //   2 MiB
    float* out = (float*)d_out;

    // one-shot packing (bf16 swizzle of weights + feature transpose)
    pack_w_kernel  <<<(8  * 4 * 512) / 256, 256, 0, stream>>>(Wb, WbS, D_, 4);
    pack_w_kernel  <<<(16 * 9 * 512) / 256, 256, 0, stream>>>(W1, W1s, C1, 9);
    pack_w_kernel  <<<(24 * 8 * 512) / 256, 256, 0, stream>>>(W2, W2s, H_, 8);
    pack_sft_kernel<<<(B_ * D_ * N_) / 256, 256, 0, stream>>>(SF, SFt);

    ball_query_kernel<<<(B_ * N_) / 8, 256, 0, stream>>>(SP, gidx);
    begin_conv_kernel<<<(512 * 2) / 8, 256, 0, stream>>>(SFt, WbS, bb, pre);
    assemble_kernel  <<<B_ * N_, 256, 0, stream>>>(SP, SFt, QP, gidx, pre, X);
    mlp1_kernel      <<<(4096 * 4) / 8, 256, 0, stream>>>(X, W1s, b1, Yb);
    mlp2_kernel      <<<(4096 * 6) / 8, 256, 0, stream>>>(Yb, W2s, b2, out);
}
